// FrameTransformerEncoder_73684458930620
// MI455X (gfx1250) — matrix operand order repaired and checked by reading, compiled, not run on hardware
//
#include <hip/hip_runtime.h>
#include <hip/hip_bf16.h>
#include <stdint.h>

// ---------------- problem constants ----------------
#define KB_   2          // batch
#define KCIN_ 2
#define KC_   4          // COUT
#define KF_   1024
#define KW_   1024
#define KH_   8          // heads
#define KD_   128        // F/HEADS
#define KFE_  4096       // F*EXP

typedef __attribute__((ext_vector_type(16))) __bf16 v16bf;
typedef __attribute__((ext_vector_type(8)))  __bf16 v8bf;
typedef __attribute__((ext_vector_type(8)))  float  v8f;

// ---------------- Tensor Data Mover support (gfx1250) ----------------
#if defined(__gfx1250__) && __has_builtin(__builtin_amdgcn_tensor_load_to_lds) && \
    __has_builtin(__builtin_amdgcn_s_wait_tensorcnt)
#define FTE_TDM 1
#else
#define FTE_TDM 0
#endif

#define FTE_AS3 __attribute__((address_space(3)))

#if FTE_TDM
typedef __attribute__((ext_vector_type(4))) unsigned fte_v4u;
typedef __attribute__((ext_vector_type(8))) int      fte_v8i;
typedef __attribute__((ext_vector_type(4))) int      fte_v4i;

// DMA a 2D tile (d0 bf16 elems per row, d1 rows, row stride = stride elems) from
// row-major global memory into LDS at byte offset lds_byte, inserting a 4-DWORD
// (8-element) pad after every row (pad_interval_code: 3 -> every 16 DW (32 elems),
// 5 -> every 64 DW (128 elems)).  One issue per wave; tracked on TENSORcnt.
__device__ __forceinline__ void fte_tdm_tile(const void* g, unsigned lds_byte,
                                             unsigned d0, unsigned d1,
                                             unsigned stride_elems,
                                             unsigned pad_interval_code) {
  unsigned long long ga = (unsigned long long)(uintptr_t)g;
  fte_v4u g0;
  g0[0] = 1u;                                            // count=1 valid descriptor
  g0[1] = lds_byte;                                      // lds_addr
  g0[2] = (unsigned)(ga & 0xffffffffu);                  // global_addr[31:0]
  g0[3] = (unsigned)((ga >> 32) & 0x1ffffffu) | (2u << 30); // addr[56:32] | type=2
  fte_v8i g1;
  // workgroup_mask=0 | data_size=1 (2B) | pad_enable | pad_interval | pad_amount=3 (4 DW)
  g1[0] = (int)((1u << 16) | (1u << 20) | (pad_interval_code << 22) | (3u << 25));
  g1[1] = (int)((stride_elems & 0xffffu) << 16);         // tensor_dim0[15:0]
  g1[2] = (int)((stride_elems >> 16) | (0xffffu << 16)); // tensor_dim0[31:16] | dim1 lo
  g1[3] = (int)(0x7fffu | (d0 << 16));                   // dim1 hi | tile_dim0
  g1[4] = (int)(d1 & 0xffffu);                           // tile_dim1 | tile_dim2=0
  g1[5] = (int)stride_elems;                             // tensor_dim0_stride[31:0]
  g1[6] = 0;                                             // stride hi | dim1_stride lo
  g1[7] = 0;
  fte_v4i g2; fte_v4i g3;
#pragma unroll
  for (int i = 0; i < 4; ++i) { g2[i] = 0; g3[i] = 0; }
#if __clang_major__ >= 23
  fte_v8i g4;
#pragma unroll
  for (int i = 0; i < 8; ++i) g4[i] = 0;
  __builtin_amdgcn_tensor_load_to_lds(g0, g1, g2, g3, g4, 0);
#else
  __builtin_amdgcn_tensor_load_to_lds(g0, g1, g2, g3, 0);
#endif
}
#endif // FTE_TDM

__device__ __forceinline__ v16bf fte_join8(v8bf lo, v8bf hi) {
  return __builtin_shufflevector(lo, hi, 0,1,2,3,4,5,6,7,8,9,10,11,12,13,14,15);
}
__device__ __forceinline__ v8f fte_wmma(v16bf a, v16bf b, v8f c) {
  return __builtin_amdgcn_wmma_f32_16x16x32_bf16(false, a, false, b, (short)0, c, false, false);
}

// ---------------- fp32 -> bf16 ----------------
__global__ void fte_cvt_bf16(const float* __restrict__ in, __bf16* __restrict__ out, size_t n) {
  size_t i = (size_t)blockIdx.x * blockDim.x + threadIdx.x;
  if (i < n) out[i] = (__bf16)in[i];
}

// ---------------- conv3x3 + bias (fp32 in, bf16 out) ----------------
__global__ void fte_conv3x3(const float* __restrict__ in, const float* __restrict__ wt,
                            const float* __restrict__ bias, __bf16* __restrict__ out,
                            int Cin, int Hs) {
  size_t idx = (size_t)blockIdx.x * blockDim.x + threadIdx.x;
  size_t total = (size_t)KB_ * KC_ * Hs * KW_;
  if (idx >= total) return;
  int x = (int)(idx % KW_);
  int y = (int)((idx / KW_) % Hs);
  int o = (int)((idx / ((size_t)KW_ * Hs)) % KC_);
  int b = (int)(idx / ((size_t)KW_ * Hs * KC_));
  float s = bias[o];
  for (int i = 0; i < Cin; ++i) {
    const float* ip = in + ((size_t)(b * Cin + i)) * Hs * KW_;
    const float* wp = wt + (o * Cin + i) * 9;
#pragma unroll
    for (int ky = 0; ky < 3; ++ky) {
      int yy = y + ky - 1;
      if ((unsigned)yy >= (unsigned)Hs) continue;
#pragma unroll
      for (int kx = 0; kx < 3; ++kx) {
        int xx = x + kx - 1;
        if ((unsigned)xx >= (unsigned)KW_) continue;
        s += wp[ky * 3 + kx] * ip[(size_t)yy * KW_ + xx];
      }
    }
  }
  out[idx] = (__bf16)s;
}

// ---------------- tiled bf16 GEMM with WMMA + TDM double buffering ----------------
// C[z][m,n] = sum_k A[z%KC_][m,k] * Bm[z][k,n]  (+ residual) (optional relu^2)
// N fixed = KW_. Block 256 thr = 8 waves, tile 128x128, wave tile 64x32.
#define LDA_PAD 40
template<int RELU2>
__global__ __launch_bounds__(256)
void fte_gemm(const __bf16* __restrict__ Aw, const __bf16* __restrict__ Bm,
              const float* __restrict__ resid, float* __restrict__ outp,
              int M, int Kdim) {
  __shared__ __bf16 sA[2][128 * LDA_PAD];
  __shared__ __bf16 sB[2][128 * LDA_PAD];
  const int N = KW_;
  const int tid  = threadIdx.x;
  const int lane = tid & 31;
  const int wave = tid >> 5;
  const int wm = wave >> 2;          // 0..1
  const int wn = wave & 3;           // 0..3
  const int n0 = blockIdx.x * 128;
  const int m0 = blockIdx.y * 128;
  const int z  = blockIdx.z;         // b*KC_ + c
  const __bf16* Ab = Aw + (size_t)(z % KC_) * M * Kdim;
  const __bf16* Bb = Bm + (size_t)z * Kdim * N;
  const size_t zoff = (size_t)z * M * N;

  v8f acc[4][2];
#pragma unroll
  for (int i = 0; i < 4; ++i)
#pragma unroll
    for (int j = 0; j < 2; ++j)
#pragma unroll
      for (int e = 0; e < 8; ++e) acc[i][j][e] = 0.f;

  // ---- staging helpers ----
  auto stageA = [&](int k0, int buf) {
#if FTE_TDM
    if (wave == 0)
      fte_tdm_tile(Ab + (size_t)m0 * Kdim + k0,
                   (unsigned)(uintptr_t)(FTE_AS3 __bf16*)&sA[buf][0],
                   /*d0=*/32u, /*d1=*/128u, (unsigned)Kdim, /*interval 16DW=*/3u);
#else
    int row = tid >> 1;
    int col = (tid & 1) * 16;
    const v8bf* g = (const v8bf*)(Ab + (size_t)(m0 + row) * Kdim + k0 + col);
    v8bf x0 = g[0], x1 = g[1];
    *(v8bf*)(&sA[buf][row * LDA_PAD + col])     = x0;
    *(v8bf*)(&sA[buf][row * LDA_PAD + col + 8]) = x1;
#endif
  };
  const int bkr = tid >> 3;
  const int bnc = (tid & 7) * 16;
  auto loadB = [&](int k0, v8bf& x0, v8bf& x1) {
    const v8bf* g = (const v8bf*)(Bb + (size_t)(k0 + bkr) * N + n0 + bnc);
    x0 = g[0]; x1 = g[1];
  };
  auto storeB = [&](int buf, v8bf x0, v8bf x1) {  // transpose -> sB[n][k]
#pragma unroll
    for (int j = 0; j < 8; ++j) sB[buf][(bnc + j) * LDA_PAD + bkr]     = x0[j];
#pragma unroll
    for (int j = 0; j < 8; ++j) sB[buf][(bnc + 8 + j) * LDA_PAD + bkr] = x1[j];
  };

  // ---- prologue: fill buffer 0 ----
  v8bf bx0, bx1;
  stageA(0, 0);
  loadB(0, bx0, bx1);
  storeB(0, bx0, bx1);
#if FTE_TDM
  if (wave == 0) __builtin_amdgcn_s_wait_tensorcnt(0);
#endif
  __syncthreads();

  for (int k0 = 0; k0 < Kdim; k0 += 32) {
    const int cur = (k0 >> 5) & 1;
    const int nxt = cur ^ 1;
    const bool more = (k0 + 32) < Kdim;
    if (more) { stageA(k0 + 32, nxt); loadB(k0 + 32, bx0, bx1); }

    v16bf afr[4];
#pragma unroll
    for (int i = 0; i < 4; ++i) {
      int row = wm * 64 + i * 16 + (lane & 15);
      int cb = (lane < 16) ? 0 : 8;   // A 16x32 layout: lanes<16 K{0..7,16..23}
      v8bf lo = *(const v8bf*)(&sA[cur][row * LDA_PAD + cb]);
      v8bf hi = *(const v8bf*)(&sA[cur][row * LDA_PAD + cb + 16]);
      afr[i] = fte_join8(lo, hi);
    }
    v16bf bfr[2];
#pragma unroll
    for (int j = 0; j < 2; ++j) {
      int coln = wn * 32 + j * 16 + (lane & 15);
      int kb = (lane < 16) ? 0 : 8;   // B 32x16 layout: lanes<16 K{0..7,16..23}
      v8bf lo = *(const v8bf*)(&sB[cur][coln * LDA_PAD + kb]);
      v8bf hi = *(const v8bf*)(&sB[cur][coln * LDA_PAD + kb + 16]);
      bfr[j] = fte_join8(lo, hi);
    }
#pragma unroll
    for (int i = 0; i < 4; ++i)
#pragma unroll
      for (int j = 0; j < 2; ++j)
        acc[i][j] = fte_wmma(afr[i], bfr[j], acc[i][j]);

    if (more) storeB(nxt, bx0, bx1);
#if FTE_TDM
    if (wave == 0) __builtin_amdgcn_s_wait_tensorcnt(0);
#endif
    __syncthreads();
  }

  // epilogue: C/D layout -> (row = r + 8*(lane>=16), col = lane&15)
#pragma unroll
  for (int i = 0; i < 4; ++i)
#pragma unroll
    for (int j = 0; j < 2; ++j) {
      int rbase = m0 + wm * 64 + i * 16 + ((lane < 16) ? 0 : 8);
      int cn = n0 + wn * 32 + j * 16 + (lane & 15);
#pragma unroll
      for (int r = 0; r < 8; ++r) {
        float v = acc[i][j][r];
        size_t off = zoff + (size_t)(rbase + r) * N + cn;
        if (resid) v += resid[off];
        if (RELU2) { float t = v > 0.f ? v : 0.f; v = t * t; }
        outp[off] = v;
      }
    }
}

// ---------------- layernorm over F axis (mcln) ----------------
__global__ void fte_mcln(const float* __restrict__ in, const float* __restrict__ g,
                         const float* __restrict__ bta, float* __restrict__ outp) {
  int w = blockIdx.y * blockDim.x + threadIdx.x;
  int z = blockIdx.x;               // b*KC_ + c
  int c = z % KC_;
  const float* base = in + (size_t)z * KF_ * KW_ + w;
  float s = 0.f, s2 = 0.f;
  for (int f = 0; f < KF_; ++f) { float v = base[(size_t)f * KW_]; s += v; s2 += v * v; }
  float mu = s * (1.f / KF_);
  float var = s2 * (1.f / KF_) - mu * mu;
  float inv = rsqrtf(var + 1e-5f);
  float* ob = outp + (size_t)z * KF_ * KW_ + w;
  for (int f = 0; f < KF_; ++f) {
    float v = (base[(size_t)f * KW_] - mu) * inv;
    ob[(size_t)f * KW_] = v * g[c * KF_ + f] + bta[c * KF_ + f];
  }
}

// ---------------- rope + head pack: (B,C,F,W) f32 -> (B,C,H,W,D) bf16 ----------------
__global__ void fte_rope_pack(const float* __restrict__ lin, __bf16* __restrict__ outp,
                              int do_rope, float scale) {
  size_t idx = (size_t)blockIdx.x * blockDim.x + threadIdx.x;
  size_t total = (size_t)KB_ * KC_ * KH_ * KW_ * (KD_ / 2);
  if (idx >= total) return;
  int d2 = (int)(idx % (KD_ / 2));
  size_t t = idx / (KD_ / 2);
  int w = (int)(t % KW_); t /= KW_;
  int h = (int)(t % KH_); t /= KH_;
  int bc = (int)t;
  size_t src = ((size_t)bc * KF_ + h * KD_ + 2 * d2) * KW_ + w;
  float x1 = lin[src];
  float x2 = lin[src + KW_];
  float e0, e1;
  if (do_rope) {
    float fr = __expf(-(float)(2 * d2) / KD_ * 9.210340371976184f); // 10000^{-2d/D}
    float ang = (float)w * fr;
    float cs = __cosf(ang), sn = __sinf(ang);
    e0 = x1 * cs - x2 * sn;
    e1 = x2 * cs + x1 * sn;
  } else { e0 = x1; e1 = x2; }
  size_t dst = (((size_t)bc * KH_ + h) * KW_ + w) * KD_ + 2 * d2;
  outp[dst]     = (__bf16)(e0 * scale);
  outp[dst + 1] = (__bf16)(e1 * scale);
}

// ---------------- flash attention ----------------
// Q,K,V: (B*C*H, W, D) bf16 (Q pre-scaled by 1/sqrt(F)); O: (B,C,F,W) fp32.
// Block 128 thr = 4 waves; wave owns 16 query rows; 32-key tiles via LDS.
#define ATW 4
__global__ __launch_bounds__(128)
void fte_attn(const __bf16* __restrict__ Q, const __bf16* __restrict__ K,
              const __bf16* __restrict__ V, float* __restrict__ O) {
  __shared__ __bf16 sK[32 * 136];          // [key][d], pad 136
  __shared__ __bf16 sV[128 * LDA_PAD];     // [d][key], pad 40
  __shared__ __bf16 sP[ATW * 16 * LDA_PAD];
  const int tid = threadIdx.x, lane = tid & 31, wave = tid >> 5;
  const int bch = blockIdx.x;              // (b*C + c)*H + hd
  const int hd = bch % KH_;
  const int bcq = bch / KH_;
  const size_t base = (size_t)bch * KW_ * KD_;
  const int w0 = blockIdx.y * (16 * ATW) + wave * 16;
  const int half = (lane < 16) ? 0 : 1;

  // preload Q fragments (A-matrix layout) for 4 k-steps of D=128
  v16bf qf[4];
  {
    int qrow = w0 + (lane & 15);
    int qoff = (lane < 16) ? 0 : 8;
#pragma unroll
    for (int c = 0; c < 4; ++c) {
      const v8bf* g = (const v8bf*)(Q + base + (size_t)qrow * KD_ + c * 32 + qoff);
      qf[c] = fte_join8(g[0], g[2]);       // +0 and +16 elements
    }
  }

  float m_run[8], l_run[8];
  v8f o_acc[8];
#pragma unroll
  for (int r = 0; r < 8; ++r) { m_run[r] = -1e30f; l_run[r] = 0.f; }
#pragma unroll
  for (int f = 0; f < 8; ++f)
#pragma unroll
    for (int e = 0; e < 8; ++e) o_acc[f][e] = 0.f;

  for (int kt = 0; kt < KW_; kt += 32) {
    __syncthreads();                       // previous-iter LDS reads complete
    // stage K tile [key][d] via TDM (fallback: manual), V tile transposed manually
#if FTE_TDM
    if (wave == 0)
      fte_tdm_tile(K + base + (size_t)kt * KD_,
                   (unsigned)(uintptr_t)(FTE_AS3 __bf16*)&sK[0],
                   /*d0=*/128u, /*d1=*/32u, /*stride=*/128u, /*interval 64DW=*/5u);
#endif
    {
      int key = tid >> 2;
      int dd = (tid & 3) * 32;
#if !FTE_TDM
      const v8bf* gk = (const v8bf*)(K + base + (size_t)(kt + key) * KD_ + dd);
      v8bf k0 = gk[0], k1 = gk[1], k2 = gk[2], k3 = gk[3];
      *(v8bf*)(&sK[key * 136 + dd])      = k0;
      *(v8bf*)(&sK[key * 136 + dd + 8])  = k1;
      *(v8bf*)(&sK[key * 136 + dd + 16]) = k2;
      *(v8bf*)(&sK[key * 136 + dd + 24]) = k3;
#endif
      const v8bf* gv = (const v8bf*)(V + base + (size_t)(kt + key) * KD_ + dd);
      v8bf va[4] = { gv[0], gv[1], gv[2], gv[3] };
#pragma unroll
      for (int q = 0; q < 4; ++q)
#pragma unroll
        for (int j = 0; j < 8; ++j) sV[(dd + q * 8 + j) * LDA_PAD + key] = va[q][j];
    }
#if FTE_TDM
    if (wave == 0) __builtin_amdgcn_s_wait_tensorcnt(0);
#endif
    __syncthreads();

    // S = Q * K^T  (16 x 32)
    v8f s[2];
#pragma unroll
    for (int j = 0; j < 2; ++j)
#pragma unroll
      for (int e = 0; e < 8; ++e) s[j][e] = 0.f;
#pragma unroll
    for (int c = 0; c < 4; ++c) {
#pragma unroll
      for (int j = 0; j < 2; ++j) {
        int keyc = j * 16 + (lane & 15);
        int kb = c * 32 + ((lane < 16) ? 0 : 8);
        v8bf lo = *(const v8bf*)(&sK[keyc * 136 + kb]);
        v8bf hi = *(const v8bf*)(&sK[keyc * 136 + kb + 16]);
        s[j] = fte_wmma(qf[c], fte_join8(lo, hi), s[j]);
      }
    }

    // online softmax (row r lives in slot r across a 16-lane group)
    float p0[8], p1[8], so[8];
#pragma unroll
    for (int r = 0; r < 8; ++r) {
      float mx = fmaxf(s[0][r], s[1][r]);
#pragma unroll
      for (int off = 8; off >= 1; off >>= 1) mx = fmaxf(mx, __shfl_xor(mx, off, 32));
      float mnew = fmaxf(m_run[r], mx);
      float sc = __expf(m_run[r] - mnew);
      float e0 = __expf(s[0][r] - mnew);
      float e1 = __expf(s[1][r] - mnew);
      float ps = e0 + e1;
#pragma unroll
      for (int off = 8; off >= 1; off >>= 1) ps += __shfl_xor(ps, off, 32);
      l_run[r] = l_run[r] * sc + ps;
      m_run[r] = mnew;
      p0[r] = e0; p1[r] = e1; so[r] = sc;
    }
#pragma unroll
    for (int f = 0; f < 8; ++f)
#pragma unroll
      for (int r = 0; r < 8; ++r) o_acc[f][r] *= so[r];

    // re-layout P (S-frag -> A-frag) through per-wave LDS
    __bf16* myP = &sP[wave * 16 * LDA_PAD];
#pragma unroll
    for (int r = 0; r < 8; ++r) {
      int row = r + 8 * half;
      myP[row * LDA_PAD + (lane & 15)]      = (__bf16)p0[r];
      myP[row * LDA_PAD + 16 + (lane & 15)] = (__bf16)p1[r];
    }
    int prow = lane & 15;
    int pc = (lane < 16) ? 0 : 8;
    v8bf plo = *(const v8bf*)(&myP[prow * LDA_PAD + pc]);
    v8bf phi = *(const v8bf*)(&myP[prow * LDA_PAD + pc + 16]);
    v16bf pf = fte_join8(plo, phi);

    // O += P * V
#pragma unroll
    for (int f = 0; f < 8; ++f) {
      int dc = f * 16 + (lane & 15);
      int kb = (lane < 16) ? 0 : 8;
      v8bf lo = *(const v8bf*)(&sV[dc * LDA_PAD + kb]);
      v8bf hi = *(const v8bf*)(&sV[dc * LDA_PAD + kb + 16]);
      o_acc[f] = fte_wmma(pf, fte_join8(lo, hi), o_acc[f]);
    }
  }

  // finalize: O[b,c, hd*D + d, w] = o_acc / l
#pragma unroll
  for (int f = 0; f < 8; ++f) {
    int d = f * 16 + (lane & 15);
#pragma unroll
    for (int r = 0; r < 8; ++r) {
      int wq = w0 + r + 8 * half;
      O[((size_t)bcq * KF_ + hd * KD_ + d) * KW_ + wq] = o_acc[f][r] / l_run[r];
    }
  }
}

// ---------------- final pack: out = [concat(x,h, axis=1), h] ----------------
__global__ void fte_pack_out(const float* __restrict__ x, const float* __restrict__ h,
                             float* __restrict__ outp) {
  size_t idx = (size_t)blockIdx.x * blockDim.x + threadIdx.x;
  const size_t FW = (size_t)KF_ * KW_;
  const size_t n1 = (size_t)KB_ * (KCIN_ + KC_) * FW;
  const size_t nh = (size_t)KB_ * KC_ * FW;
  if (idx < n1) {
    size_t rest = idx % FW;
    size_t ch = (idx / FW) % (KCIN_ + KC_);
    size_t b = idx / (FW * (KCIN_ + KC_));
    outp[idx] = (ch < KCIN_) ? x[(b * KCIN_ + ch) * FW + rest]
                             : h[(b * KC_ + (ch - KCIN_)) * FW + rest];
  } else if (idx < n1 + nh) {
    outp[idx] = h[idx - n1];
  }
}

// ---------------- orchestration ----------------
extern "C" void kernel_launch(void* const* d_in, const int* in_sizes, int n_in,
                              void* d_out, int out_size, void* d_ws, size_t ws_size,
                              hipStream_t stream) {
  const float* x        = (const float*)d_in[0];
  const float* embed_cw = (const float*)d_in[1];
  const float* embed_cb = (const float*)d_in[2];
  const float* embed_pw = (const float*)d_in[3];
  const float* norm1_g  = (const float*)d_in[4];
  const float* norm1_b  = (const float*)d_in[5];
  const float* q_cw = (const float*)d_in[6],  *q_cb = (const float*)d_in[7],  *q_pw = (const float*)d_in[8];
  const float* k_cw = (const float*)d_in[9],  *k_cb = (const float*)d_in[10], *k_pw = (const float*)d_in[11];
  const float* v_cw = (const float*)d_in[12], *v_cb = (const float*)d_in[13], *v_pw = (const float*)d_in[14];
  const float* o_cw = (const float*)d_in[15], *o_cb = (const float*)d_in[16], *o_pw = (const float*)d_in[17];
  const float* norm2_g = (const float*)d_in[18];
  const float* norm2_b = (const float*)d_in[19];
  const float* c1_cw = (const float*)d_in[20], *c1_cb = (const float*)d_in[21], *c1_pw = (const float*)d_in[22];
  const float* c2_cw = (const float*)d_in[23], *c2_cb = (const float*)d_in[24], *c2_pw = (const float*)d_in[25];

  const size_t BCFW  = (size_t)KB_ * KC_ * KF_ * KW_;   // 8M
  const size_t WSQ   = (size_t)KC_ * KF_ * KF_;         // 4M  (per-channel FxF)
  const size_t WBIG  = (size_t)KC_ * KFE_ * KF_;        // 16M
  const size_t BIGA  = (size_t)KB_ * KC_ * KFE_ * KW_;  // 32M

  char* p = (char*)d_ws;
  auto alloc = [&](size_t bytes) { char* r = p; p += (bytes + 255) & ~(size_t)255; return r; };
  __bf16* w_embed = (__bf16*)alloc(WSQ * 2);
  __bf16* w_q     = (__bf16*)alloc(WSQ * 2);
  __bf16* w_k     = (__bf16*)alloc(WSQ * 2);
  __bf16* w_v     = (__bf16*)alloc(WSQ * 2);
  __bf16* w_o     = (__bf16*)alloc(WSQ * 2);
  __bf16* w_c1    = (__bf16*)alloc(WBIG * 2);
  __bf16* w_c2    = (__bf16*)alloc(WBIG * 2);
  __bf16* conv_bf = (__bf16*)alloc(BCFW * 2);
  __bf16* convbig = (__bf16*)alloc(BIGA * 2);
  float*  hbuf    = (float*)alloc(BCFW * 4);
  float*  hn      = (float*)alloc(BCFW * 4);
  float*  lin     = (float*)alloc(BCFW * 4);   // reused: q/k/v lin, attn out
  float*  bigf    = (float*)alloc(BIGA * 4);
  __bf16* q_bf    = (__bf16*)alloc(BCFW * 2);
  __bf16* k_bf    = (__bf16*)alloc(BCFW * 2);
  __bf16* v_bf    = (__bf16*)alloc(BCFW * 2);

  auto gcvt = [&](const float* src, __bf16* dst, size_t n) {
    fte_cvt_bf16<<<(unsigned)((n + 255) / 256), 256, 0, stream>>>(src, dst, n);
  };
  gcvt(embed_pw, w_embed, WSQ);
  gcvt(q_pw, w_q, WSQ); gcvt(k_pw, w_k, WSQ); gcvt(v_pw, w_v, WSQ); gcvt(o_pw, w_o, WSQ);
  gcvt(c1_pw, w_c1, WBIG); gcvt(c2_pw, w_c2, WBIG);

  const unsigned convGrid    = (unsigned)((BCFW + 255) / 256);
  const unsigned convGridBig = (unsigned)((BIGA + 255) / 256);
  dim3 gemmGrid(KW_ / 128, KF_ / 128, KB_ * KC_);       // M = 1024
  dim3 gemmGridC1(KW_ / 128, KFE_ / 128, KB_ * KC_);    // M = 4096
  dim3 lnGrid(KB_ * KC_, KW_ / 256);
  const unsigned ropeGrid = (unsigned)(((size_t)KB_ * KC_ * KH_ * KW_ * (KD_ / 2) + 255) / 256);
  dim3 attnGrid(KB_ * KC_ * KH_, KW_ / (16 * ATW));

  // h = mlinear(x, embed)
  fte_conv3x3<<<convGrid, 256, 0, stream>>>(x, embed_cw, embed_cb, conv_bf, KCIN_, KF_);
  fte_gemm<0><<<gemmGrid, 256, 0, stream>>>(w_embed, conv_bf, nullptr, hbuf, KF_, KF_);
  // hn = mcln(h)
  fte_mcln<<<lnGrid, 256, 0, stream>>>(hbuf, norm1_g, norm1_b, hn);
  // q/k/v projections (+rope, q scaled by 1/sqrt(F)=1/32)
  fte_conv3x3<<<convGrid, 256, 0, stream>>>(hn, q_cw, q_cb, conv_bf, KC_, KF_);
  fte_gemm<0><<<gemmGrid, 256, 0, stream>>>(w_q, conv_bf, nullptr, lin, KF_, KF_);
  fte_rope_pack<<<ropeGrid, 256, 0, stream>>>(lin, q_bf, 1, 1.0f / 32.0f);
  fte_conv3x3<<<convGrid, 256, 0, stream>>>(hn, k_cw, k_cb, conv_bf, KC_, KF_);
  fte_gemm<0><<<gemmGrid, 256, 0, stream>>>(w_k, conv_bf, nullptr, lin, KF_, KF_);
  fte_rope_pack<<<ropeGrid, 256, 0, stream>>>(lin, k_bf, 1, 1.0f);
  fte_conv3x3<<<convGrid, 256, 0, stream>>>(hn, v_cw, v_cb, conv_bf, KC_, KF_);
  fte_gemm<0><<<gemmGrid, 256, 0, stream>>>(w_v, conv_bf, nullptr, lin, KF_, KF_);
  fte_rope_pack<<<ropeGrid, 256, 0, stream>>>(lin, v_bf, 0, 1.0f);
  // attention -> lin (as (B,C,F,W) fp32)
  fte_attn<<<attnGrid, 128, 0, stream>>>(q_bf, k_bf, v_bf, lin);
  // h += mlinear(attn, o)
  fte_conv3x3<<<convGrid, 256, 0, stream>>>(lin, o_cw, o_cb, conv_bf, KC_, KF_);
  fte_gemm<0><<<gemmGrid, 256, 0, stream>>>(w_o, conv_bf, hbuf, hbuf, KF_, KF_);
  // z = mcln(h); z1 = relu(mlinear(z, c1))^2
  fte_mcln<<<lnGrid, 256, 0, stream>>>(hbuf, norm2_g, norm2_b, hn);
  fte_conv3x3<<<convGrid, 256, 0, stream>>>(hn, c1_cw, c1_cb, conv_bf, KC_, KF_);
  fte_gemm<1><<<gemmGridC1, 256, 0, stream>>>(w_c1, conv_bf, nullptr, bigf, KFE_, KF_);
  // h += mlinear(z1, c2)
  fte_conv3x3<<<convGridBig, 256, 0, stream>>>(bigf, c2_cw, c2_cb, convbig, KC_, KFE_);
  fte_gemm<0><<<gemmGrid, 256, 0, stream>>>(w_c2, convbig, hbuf, hbuf, KF_, KFE_);
  // out = [concat(x,h), h]
  const size_t nout = (size_t)KB_ * (KCIN_ + KC_) * KF_ * KW_ + BCFW;
  fte_pack_out<<<(unsigned)((nout + 255) / 256), 256, 0, stream>>>(x, hbuf, (float*)d_out);
  (void)in_sizes; (void)n_in; (void)out_size; (void)ws_size;
}
